// TransitionUp2_52372831207678
// MI455X (gfx1250) — matrix operand order repaired and checked by reading, not compiled, not run on hardware
//
#include <hip/hip_runtime.h>
#include <cstddef>

// ---------------------------------------------------------------------------
// Problem constants (match reference)
// ---------------------------------------------------------------------------
#define NPTS   32768
#define NSUB   8192
#define INC    256
#define OUTC   128
#define KNN    8
#define BN_EPS 1e-5f

typedef __attribute__((ext_vector_type(16))) __bf16 v16bf;
typedef __attribute__((ext_vector_type(8)))  float  v8f;
typedef unsigned int u32x4 __attribute__((ext_vector_type(4)));
typedef int          i32x4 __attribute__((ext_vector_type(4)));
typedef int          i32x8 __attribute__((ext_vector_type(8)));

#define AS1 __attribute__((address_space(1)))
#define AS3 __attribute__((address_space(3)))

__device__ __forceinline__ float silu(float v) { return v / (1.0f + __expf(-v)); }

// pack two floats into one dword of two bf16 (lo = a, hi = b)
__device__ __forceinline__ unsigned int pack_bf2(float a, float b) {
    union { __bf16 h[2]; unsigned int u; } u;
    u.h[0] = (__bf16)a; u.h[1] = (__bf16)b;
    return u.u;
}

// Fragment-major swizzle maps (CDNA5 ISA 7.12.2, wave32, 16x16x32 bf16):
//  A 16x32: element(row rr, K) -> lane = rr + 16*(K>>4),     slot = K & 15
//  B 32x16: element(K, col cc) -> lane = cc + 16*(K>>4),     slot = K & 15
// Each lane's 16 bf16 operand values are contiguous -> two ds_load_b128.

// ---------------------------------------------------------------------------
// Kernel 1: brute-force kNN. pos_sub (8192x3 f32 = 96KB) moved to LDS with
// the Tensor Data Mover (one DMA descriptor issued by wave 0, completion via
// s_wait_tensorcnt), falling back to async global->LDS b128 copies, then to
// plain copies. One thread per target, per-thread sorted top-8 registers.
// ---------------------------------------------------------------------------
__global__ __launch_bounds__(256) void knn_kernel(
    const float* __restrict__ pos, const float* __restrict__ pos_sub,
    int* __restrict__ idx)
{
    extern __shared__ char dynsmem[];
    float* sp = (float*)dynsmem;                       // NSUB*3 floats (96KB)

#if __has_builtin(__builtin_amdgcn_tensor_load_to_lds)
    // ---- TDM path: one 1-row tile of 24576 x 4B elements, global -> LDS ----
    if (threadIdx.x < 32) {                            // wave 0 issues the DMA
        const unsigned long long gaddr = (unsigned long long)pos_sub;
        const unsigned lds_base =
            (unsigned)(unsigned long long)(AS3 char*)dynsmem;
        u32x4 g0;
        g0.x = 1u;                                     // count=1, user mode
        g0.y = lds_base;                               // lds_addr (bytes)
        g0.z = (unsigned)(gaddr & 0xFFFFFFFFu);        // global_addr[31:0]
        g0.w = (unsigned)((gaddr >> 32) & 0x01FFFFFFu) // global_addr[56:32]
             | (2u << 30);                             // type = 2 ("image")
        i32x8 g1;
        g1[0] = 0x00020000;                            // data_size = 2 (4B)
        g1[1] = (int)(24576u << 16);                   // tensor_dim0 = 24576 (lo16)
        g1[2] = 0x00010000;                            // tensor_dim1 = 1
        g1[3] = (int)(24576u << 16);                   // tile_dim0 = 24576
        g1[4] = 1;                                     // tile_dim1 = 1
        g1[5] = 24576;                                 // tensor_dim0_stride
        g1[6] = 0;
        g1[7] = 0;
        const i32x4 z4 = {0, 0, 0, 0};
#if __clang_major__ >= 23
        const i32x8 z8 = {0, 0, 0, 0, 0, 0, 0, 0};
        __builtin_amdgcn_tensor_load_to_lds(g0, g1, z4, z4, z8, 0);
#else
        __builtin_amdgcn_tensor_load_to_lds(g0, g1, z4, z4, 0);
#endif
        __builtin_amdgcn_s_wait_tensorcnt(0);
    }
#elif __has_builtin(__builtin_amdgcn_global_load_async_to_lds_b128)
    for (int i = threadIdx.x * 4; i < NSUB * 3; i += 256 * 4) {
        __builtin_amdgcn_global_load_async_to_lds_b128(
            (AS1 void*)(pos_sub + i), (AS3 void*)(sp + i), 0, 0);
    }
#if __has_builtin(__builtin_amdgcn_s_wait_asynccnt)
    __builtin_amdgcn_s_wait_asynccnt(0);
#else
    asm volatile("s_wait_asynccnt 0" ::: "memory");
#endif
#else
    for (int i = threadIdx.x; i < NSUB * 3; i += 256) sp[i] = pos_sub[i];
#endif
    __syncthreads();

    const int t  = blockIdx.x * 256 + threadIdx.x;
    const float px = pos[t * 3 + 0];
    const float py = pos[t * 3 + 1];
    const float pz = pos[t * 3 + 2];

    float bd[KNN];
    int   bi[KNN];
#pragma unroll
    for (int k = 0; k < KNN; ++k) { bd[k] = 3.0e38f; bi[k] = 0; }

    for (int j = 0; j < NSUB; ++j) {
        const float dx = px - sp[3 * j + 0];
        const float dy = py - sp[3 * j + 1];
        const float dz = pz - sp[3 * j + 2];
        const float d  = dx * dx + dy * dy + dz * dz;
        if (d < bd[KNN - 1]) {
            bd[KNN - 1] = d; bi[KNN - 1] = j;
#pragma unroll
            for (int k = KNN - 1; k > 0; --k) {
                if (bd[k] < bd[k - 1]) {
                    float td = bd[k]; bd[k] = bd[k - 1]; bd[k - 1] = td;
                    int   ti = bi[k]; bi[k] = bi[k - 1]; bi[k - 1] = ti;
                }
            }
        }
    }
#pragma unroll
    for (int k = 0; k < KNN; ++k) idx[t * KNN + k] = bi[k];
}

// ---------------------------------------------------------------------------
// Kernel 2: C[M,128] = A[M,K] @ B[K,128] (f32 in/out, bf16 WMMA, f32 acc).
// Block = 256 threads = 8 waves; block tile = 64x128; wave = 16x64 strip.
// Double-buffered LDS tiles in fragment-major layout; all four B fragments
// are loaded into distinct registers before the WMMA chain so the ds_load
// latency overlaps the first matrix ops (partial s_wait_dscnt).
// epilogue: 0 = plain, 1 = +bias, BatchNorm(eval), SiLU (self.mlp path).
// ---------------------------------------------------------------------------
__global__ __launch_bounds__(256) void gemm_bf16_kernel(
    const float* __restrict__ A, const float* __restrict__ B,
    float* __restrict__ C, int M, int Kdim, int epilogue,
    const float* __restrict__ bias, const float* __restrict__ gamma,
    const float* __restrict__ beta, const float* __restrict__ rmean,
    const float* __restrict__ rvar)
{
    extern __shared__ char dynsmem[];  // 2 x (Af 4KB + Bf 8KB) = 24KB

    const int tid  = threadIdx.x;
    const int wave = tid >> 5;
    const int lane = tid & 31;
    const int mBlk = blockIdx.x * 64;
    const int mt   = wave & 3;          // m-tile of this wave (16 rows)
    const int ntb  = (wave >> 2) * 4;   // first n-tile of this wave (of 8)
    const int h    = lane >> 4;
    const int cc   = lane & 15;
    const int steps = Kdim >> 5;

    float2 aR[4];
    float  bR[16];

    auto loadRegs = [&](int s) {
        const int k0 = s << 5;
#pragma unroll
        for (int u = 0; u < 4; ++u) {
            const int i = tid + 256 * u;
            const int r = i >> 4, k = (i & 15) * 2;
            aR[u] = *(const float2*)(A + (size_t)(mBlk + r) * Kdim + k0 + k);
        }
#pragma unroll
        for (int u = 0; u < 8; ++u) {
            const int i = tid + 256 * u;
            const int c = i & 127, k = (i >> 7) * 2;
            bR[2 * u]     = B[(size_t)(k0 + k) * OUTC + c];
            bR[2 * u + 1] = B[(size_t)(k0 + k + 1) * OUTC + c];
        }
    };
    auto storeLDS = [&](int b) {
        unsigned int* Af = (unsigned int*)(dynsmem + b * 12288);
        unsigned int* Bf = (unsigned int*)(dynsmem + b * 12288 + 4096);
#pragma unroll
        for (int u = 0; u < 4; ++u) {
            const int i = tid + 256 * u;
            const int r = i >> 4, k = (i & 15) * 2;
            const int ln = (r & 15) + 16 * (k >> 4);
            const int sl = k & 15;
            Af[(((r >> 4) * 32 + ln) * 16 + sl) >> 1] = pack_bf2(aR[u].x, aR[u].y);
        }
#pragma unroll
        for (int u = 0; u < 8; ++u) {
            const int i = tid + 256 * u;
            const int c = i & 127, k = (i >> 7) * 2;
            const int ln = (c & 15) + 16 * (k >> 4);
            const int sl = k & 15;
            Bf[(((c >> 4) * 32 + ln) * 16 + sl) >> 1] = pack_bf2(bR[2 * u], bR[2 * u + 1]);
        }
    };

    loadRegs(0);
    storeLDS(0);
    __syncthreads();

    v8f acc[4] = {};
    for (int s = 0; s < steps; ++s) {
        const int cur = s & 1;
        if (s + 1 < steps) loadRegs(s + 1);      // overlap next-tile loads
        const __bf16* Af = (const __bf16*)(dynsmem + cur * 12288);
        const __bf16* Bf = (const __bf16*)(dynsmem + cur * 12288 + 4096);
        // issue ALL fragment loads first (distinct registers), then the WMMA
        // chain -> partial dscnt waits instead of a full drain per WMMA
        const v16bf af = *(const v16bf*)(Af + (mt * 32 + lane) * 16);
        v16bf bfr[4];
#pragma unroll
        for (int t = 0; t < 4; ++t)
            bfr[t] = *(const v16bf*)(Bf + ((ntb + t) * 32 + lane) * 16);
#pragma unroll
        for (int t = 0; t < 4; ++t)
            acc[t] = __builtin_amdgcn_wmma_f32_16x16x32_bf16(
                false, af, false, bfr[t], (short)0, acc[t], false, false);
        if (s + 1 < steps) {
            storeLDS(cur ^ 1);
            __syncthreads();
        }
    }

#pragma unroll
    for (int t = 0; t < 4; ++t) {
        const int n0 = (ntb + t) * 16;
#pragma unroll
        for (int r = 0; r < 8; ++r) {
            const int row = mBlk + mt * 16 + r + 8 * h;
            const int col = n0 + cc;
            float v = acc[t][r];
            if (epilogue == 1) {
                v += bias[col];
                v = (v - rmean[col]) * rsqrtf(rvar[col] + BN_EPS) * gamma[col] + beta[col];
                v = silu(v);
            }
            C[(size_t)row * OUTC + col] = v;
        }
    }
}

// ---------------------------------------------------------------------------
// Kernel 3: fused PointTransformerConv message pass.
// Block = 128 threads = 4 waves; wave owns 2 targets (16 edges = one WMMA
// M-tile). Wa pre-swizzled once per block into fragment-major LDS; E written
// pair-wise directly in fragment layout (channel == GEMM K dimension).
// All 4 k-tile fragments loaded up front before the WMMA chain.
// Softmax over K=8 is per-lane across the 8 accumulator VGPRs (C layout puts
// all 8 edges of one target in one lane). Adds H = silu(BN(x@Wm+bm)).
// ---------------------------------------------------------------------------
__global__ __launch_bounds__(128) void message_kernel(
    const float* __restrict__ pos, const float* __restrict__ pos_sub,
    const int* __restrict__ idx,
    const float* __restrict__ Adst, const float* __restrict__ Asrc,
    const float* __restrict__ Vsrc,
    const float* __restrict__ Wa, const float* __restrict__ ba,
    const float* __restrict__ Wp, const float* __restrict__ bp,
    const float* __restrict__ H, float* __restrict__ out)
{
    extern __shared__ char dynsmem[];
    // layout: WaF 32KB | EF 16KB (4KB/wave) | VD 32KB (8KB/wave) | consts 2.5KB
    __bf16* sWaF = (__bf16*)dynsmem;                       // [kt][nt][lane][16]
    __bf16* sEF  = (__bf16*)(dynsmem + 32768);             // per wave [kt][lane][16]
    float*  sVD  = (float*)(dynsmem + 32768 + 16384);      // per wave [16][128]
    float*  sba  = (float*)(dynsmem + 32768 + 16384 + 32768);
    float*  sWp  = sba + 128;                              // 3*128
    float*  sbp  = sWp + 384;                              // 128

    const int tid  = threadIdx.x;
    const int wave = tid >> 5;
    const int lane = tid & 31;

    // ---- stage Wa fragment-major (whole block): pairs along K ----
    for (int i = tid; i < 64 * 128; i += 128) {
        const int c = i & 127;
        const int k = (i >> 7) * 2;     // even K, 0..126
        const float w0 = Wa[(size_t)k * OUTC + c];
        const float w1 = Wa[(size_t)(k + 1) * OUTC + c];
        const int kt = k >> 5, kk = k & 31;
        const int ln = (c & 15) + 16 * (kk >> 4);
        const int sl = kk & 15;
        ((unsigned int*)sWaF)[((((kt * 8) + (c >> 4)) * 32 + ln) * 16 + sl) >> 1] =
            pack_bf2(w0, w1);
    }
    if (tid < 128) { sba[tid] = ba[tid]; sbp[tid] = bp[tid]; }
    for (int i = tid; i < 384; i += 128) sWp[i] = Wp[i];
    __syncthreads();

    // ---- per-wave: stage E (fragment-major bf16) and VD (row-major f32) ----
    const int t0 = blockIdx.x * 8 + wave * 2;
    __bf16* mEF = sEF + wave * (4 * 32 * 16);
    float*  mVD = sVD + wave * (16 * 128);

    for (int ed = 0; ed < 16; ++ed) {
        const int tg = t0 + (ed >> 3);
        const int j  = idx[tg * KNN + (ed & 7)];
        const float dpx = pos[tg * 3 + 0] - pos_sub[j * 3 + 0];
        const float dpy = pos[tg * 3 + 1] - pos_sub[j * 3 + 1];
        const float dpz = pos[tg * 3 + 2] - pos_sub[j * 3 + 2];
#pragma unroll
        for (int it = 0; it < 2; ++it) {
            const int c = 2 * lane + 64 * it;                  // even channel
            const float2 ad = *(const float2*)(Adst + (size_t)tg * OUTC + c);
            const float2 as = *(const float2*)(Asrc + (size_t)j  * OUTC + c);
            const float2 vs = *(const float2*)(Vsrc + (size_t)j  * OUTC + c);
            const float dl0 = silu(dpx * sWp[c]     + dpy * sWp[128 + c]     +
                                   dpz * sWp[256 + c]     + sbp[c]);
            const float dl1 = silu(dpx * sWp[c + 1] + dpy * sWp[128 + c + 1] +
                                   dpz * sWp[256 + c + 1] + sbp[c + 1]);
            const float e0 = ad.x - as.x + dl0;
            const float e1 = ad.y - as.y + dl1;
            // E fragment-major: row = edge, K = channel
            const int kt = c >> 5, kk = c & 31;
            const int ln = ed + 16 * (kk >> 4);
            const int sl = kk & 15;
            ((unsigned int*)mEF)[((kt * 32 + ln) * 16 + sl) >> 1] = pack_bf2(e0, e1);
            float2 vd; vd.x = vs.x + dl0; vd.y = vs.y + dl1;
            *(float2*)(mVD + ed * 128 + c) = vd;
        }
    }
    __syncthreads();

    // ---- per-wave: attention matmul + per-lane softmax + weighted sum ----
    const int h   = lane >> 4;
    const int cc  = lane & 15;
    const int tgt = t0 + h;            // lanes 0-15 -> t0, lanes 16-31 -> t0+1

    // A fragments (one per k-tile) are invariant across n-tiles: load once.
    v16bf afk[4];
#pragma unroll
    for (int kt = 0; kt < 4; ++kt)
        afk[kt] = *(const v16bf*)(mEF + (kt * 32 + lane) * 16);

    for (int nt = 0; nt < 8; ++nt) {
        v16bf bfk[4];
#pragma unroll
        for (int kt = 0; kt < 4; ++kt)
            bfk[kt] = *(const v16bf*)(sWaF + ((kt * 8 + nt) * 32 + lane) * 16);
        v8f acc = {};
#pragma unroll
        for (int kt = 0; kt < 4; ++kt)
            acc = __builtin_amdgcn_wmma_f32_16x16x32_bf16(
                false, afk[kt], false, bfk[kt], (short)0, acc, false, false);

        const int col = nt * 16 + cc;
        // acc[r] = attn logit of edge (r + 8h) for channel `col`
        float a[8];
        float mx = -3.0e38f;
#pragma unroll
        for (int r = 0; r < 8; ++r) {
            const float v = silu(acc[r] + sba[col]);
            a[r] = v;
            mx = fmaxf(mx, v);
        }
        float s = 0.0f;
#pragma unroll
        for (int r = 0; r < 8; ++r) { a[r] = __expf(a[r] - mx); s += a[r]; }
        const float inv = 1.0f / s;
        float xi = 0.0f;
#pragma unroll
        for (int r = 0; r < 8; ++r) {
            xi += a[r] * inv * mVD[(r + 8 * h) * 128 + col];
        }
        out[(size_t)tgt * OUTC + col] = H[(size_t)tgt * OUTC + col] + xi;
    }
}

// ---------------------------------------------------------------------------
// Host launcher
// ---------------------------------------------------------------------------
extern "C" void kernel_launch(void* const* d_in, const int* in_sizes, int n_in,
                              void* d_out, int out_size, void* d_ws, size_t ws_size,
                              hipStream_t stream)
{
    (void)in_sizes; (void)n_in; (void)out_size; (void)ws_size;

    const float* x       = (const float*)d_in[0];
    const float* x_sub   = (const float*)d_in[1];
    const float* pos     = (const float*)d_in[2];
    const float* pos_sub = (const float*)d_in[3];
    const float* Wlin    = (const float*)d_in[4];
    const float* Wsrc    = (const float*)d_in[5];
    const float* Wdst    = (const float*)d_in[6];
    const float* Wp      = (const float*)d_in[7];
    const float* bp      = (const float*)d_in[8];
    const float* Wa      = (const float*)d_in[9];
    const float* ba      = (const float*)d_in[10];
    const float* Wm      = (const float*)d_in[11];
    const float* bm      = (const float*)d_in[12];
    const float* gamma   = (const float*)d_in[13];
    const float* beta    = (const float*)d_in[14];
    const float* rmean   = (const float*)d_in[15];
    const float* rvar    = (const float*)d_in[16];
    float* out = (float*)d_out;

    // scratch layout
    char* ws = (char*)d_ws;
    int*   idx  = (int*)ws;   ws += (size_t)NPTS * KNN  * sizeof(int);    // 1 MB
    float* Vv   = (float*)ws; ws += (size_t)NSUB * OUTC * sizeof(float);  // 4 MB
    float* Av   = (float*)ws; ws += (size_t)NSUB * OUTC * sizeof(float);  // 4 MB
    float* Ad   = (float*)ws; ws += (size_t)NPTS * OUTC * sizeof(float);  // 16 MB
    float* Hh   = (float*)ws; ws += (size_t)NPTS * OUTC * sizeof(float);  // 16 MB

    const int LDS_KNN  = NSUB * 3 * (int)sizeof(float);                   // 96 KB
    const int LDS_GEMM = 2 * (4096 + 8192);                               // 24 KB
    const int LDS_MSG  = 32768 + 16384 + 32768 + (128 + 384 + 128) * 4;   // ~84 KB

    hipFuncSetAttribute((const void*)knn_kernel,
                        hipFuncAttributeMaxDynamicSharedMemorySize, LDS_KNN);
    hipFuncSetAttribute((const void*)gemm_bf16_kernel,
                        hipFuncAttributeMaxDynamicSharedMemorySize, LDS_GEMM);
    hipFuncSetAttribute((const void*)message_kernel,
                        hipFuncAttributeMaxDynamicSharedMemorySize, LDS_MSG);

    // 1) kNN (pos_sub -> LDS via TDM / async-LDS path)
    knn_kernel<<<NPTS / 256, 256, LDS_KNN, stream>>>(pos, pos_sub, idx);

    // 2) GEMMs (bf16 WMMA, f32 accumulate, double-buffered LDS)
    gemm_bf16_kernel<<<NSUB / 64, 256, LDS_GEMM, stream>>>(
        x_sub, Wlin, Vv, NSUB, INC, 0, nullptr, nullptr, nullptr, nullptr, nullptr);
    gemm_bf16_kernel<<<NSUB / 64, 256, LDS_GEMM, stream>>>(
        x_sub, Wsrc, Av, NSUB, INC, 0, nullptr, nullptr, nullptr, nullptr, nullptr);
    gemm_bf16_kernel<<<NPTS / 64, 256, LDS_GEMM, stream>>>(
        x, Wdst, Ad, NPTS, OUTC, 0, nullptr, nullptr, nullptr, nullptr, nullptr);
    gemm_bf16_kernel<<<NPTS / 64, 256, LDS_GEMM, stream>>>(
        x, Wm, Hh, NPTS, OUTC, 1, bm, gamma, beta, rmean, rvar);

    // 3) fused attention message pass (+ residual add of H) -> d_out
    message_kernel<<<NPTS / 8, 128, LDS_MSG, stream>>>(
        pos, pos_sub, idx, Ad, Av, Vv, Wa, ba, Wp, bp, Hh, out);
}
